// VectorQuantize_66606352827342
// MI455X (gfx1250) — matrix operand order repaired and checked by reading, not compiled, not run on hardware
//
#include <hip/hip_runtime.h>

typedef __attribute__((ext_vector_type(16))) __bf16 v16bf;
typedef __attribute__((ext_vector_type(8)))  __bf16 v8bf;
typedef __attribute__((ext_vector_type(8)))  float  v8f;
typedef __attribute__((ext_vector_type(4)))  unsigned int u32x4;
typedef __attribute__((ext_vector_type(4)))  int i32x4;
typedef __attribute__((ext_vector_type(8)))  int i32x8;

#define DIM            512
#define KCODES         8192
#define ROWS_PER_BLOCK 64
#define COLS_PER_STEP  32
#define KCHUNKS        (DIM / 32)          // 16 WMMA k-steps
#define NTILES         (KCODES / COLS_PER_STEP)

__device__ __forceinline__ unsigned short f32_to_bf16_rne(float f) {
  unsigned int u = __float_as_uint(f);
  unsigned int r = u + 0x7FFFu + ((u >> 16) & 1u);
  return (unsigned short)(r >> 16);
}

__device__ __forceinline__ unsigned pack_bf16x2_rne(float a, float b) {
  unsigned ua = __float_as_uint(a), ub = __float_as_uint(b);
  unsigned ra = ua + 0x7FFFu + ((ua >> 16) & 1u);
  unsigned rb = ub + 0x7FFFu + ((ub >> 16) & 1u);
  return (ra >> 16) | (rb & 0xFFFF0000u);
}

// Low 32 bits of a generic LDS pointer == wave-relative LDS byte offset.
__device__ __forceinline__ unsigned lds_off(const void* p) {
  return (unsigned)(unsigned long long)p;
}

// ---------------- Kernel 1: e_sq[k] = sum(embed[k,:]^2), one wave32 per row ---
__global__ __launch_bounds__(256) void vq_esq_kernel(const float* __restrict__ embed,
                                                     float* __restrict__ esq) {
  const int lane = threadIdx.x & 31;
  const int wave = threadIdx.x >> 5;
  const int row  = blockIdx.x * 8 + wave;
  if (row >= KCODES) return;
  const float* e = embed + (size_t)row * DIM;
  float s = 0.0f;
  for (int d = lane; d < DIM; d += 32) { float v = e[d]; s += v * v; }
  for (int off = 16; off > 0; off >>= 1) s += __shfl_xor(s, off, 32);
  if (lane == 0) esq[row] = s;
}

// ---------------- Kernel 1b: embed fp32 -> bf16 workspace ---------------------
__global__ __launch_bounds__(256) void vq_embed_bf16_kernel(const float* __restrict__ embed,
                                                            unsigned short* __restrict__ ebf) {
  const int i = blockIdx.x * 256 + threadIdx.x;      // float4 index
  float4 v = ((const float4*)embed)[i];
  uint2 p;
  p.x = pack_bf16x2_rne(v.x, v.y);
  p.y = pack_bf16x2_rne(v.z, v.w);
  ((uint2*)ebf)[i] = p;
}

// ---- shared helpers ----------------------------------------------------------
union AU { v16bf v; unsigned short s[16]; };

// A fragment, contiguous k order (B matches; not the ISA table order): lanes 0-15 hold K {0..15},
// lanes 16-31 hold K {16..31} of row l16.
__device__ __forceinline__ void load_A_fragment(AU* A, const float* gx, int half) {
#pragma unroll
  for (int i = 0; i < KCHUNKS; ++i) {
    const float* p0 = gx + 32 * i + half * 16;
    const float* p1 = p0 + 8;
    float4 q0 = *(const float4*)(p0);
    float4 q1 = *(const float4*)(p0 + 4);
    float4 q2 = *(const float4*)(p1);
    float4 q3 = *(const float4*)(p1 + 4);
    A[i].s[0]  = f32_to_bf16_rne(q0.x);  A[i].s[1]  = f32_to_bf16_rne(q0.y);
    A[i].s[2]  = f32_to_bf16_rne(q0.z);  A[i].s[3]  = f32_to_bf16_rne(q0.w);
    A[i].s[4]  = f32_to_bf16_rne(q1.x);  A[i].s[5]  = f32_to_bf16_rne(q1.y);
    A[i].s[6]  = f32_to_bf16_rne(q1.z);  A[i].s[7]  = f32_to_bf16_rne(q1.w);
    A[i].s[8]  = f32_to_bf16_rne(q2.x);  A[i].s[9]  = f32_to_bf16_rne(q2.y);
    A[i].s[10] = f32_to_bf16_rne(q2.z);  A[i].s[11] = f32_to_bf16_rne(q2.w);
    A[i].s[12] = f32_to_bf16_rne(q3.x);  A[i].s[13] = f32_to_bf16_rne(q3.y);
    A[i].s[14] = f32_to_bf16_rne(q3.z);  A[i].s[15] = f32_to_bf16_rne(q3.w);
  }
}

// TDM: one descriptor-driven DMA moves a whole 32x512 bf16 tile to LDS.
// D# layout per CDNA5 ISA §8.3/8.4 (2D tensor; groups 2/3 zero).
__device__ __forceinline__ void tdm_load_tile(const unsigned short* ebf, int tile,
                                              unsigned lds_addr) {
  unsigned long long ga =
      (unsigned long long)(const void*)(ebf + (size_t)tile * COLS_PER_STEP * DIM);
  u32x4 g0;
  g0[0] = 1u;                                             // count=1 (valid user D#)
  g0[1] = lds_addr;                                       // lds_addr [63:32]
  g0[2] = (unsigned)ga;                                   // global_addr lo
  g0[3] = (unsigned)((ga >> 32) & 0x01FFFFFFull)          // global_addr [56:32]
          | (2u << 30);                                   // type=2 ("image")
  i32x8 g1;
  g1[0] = 0x00010000;                                     // data_size=1 (2B), no pad/mcast
  g1[1] = (int)(DIM << 16);                               // tensor_dim0 = 512
  g1[2] = (int)(KCODES << 16);                            // tensor_dim1 = 8192 (lo16)
  g1[3] = (int)(DIM << 16);                               // tile_dim0 = 512 (dim1 hi16 = 0)
  g1[4] = COLS_PER_STEP;                                  // tile_dim1 = 32, tile_dim2 = 0
  g1[5] = DIM;                                            // tensor_dim0_stride = 512
  g1[6] = 0;                                              // stride hi / dim1_stride lo
  g1[7] = 0;
  i32x4 z4 = {0, 0, 0, 0};
#if __clang_major__ >= 23
  i32x8 z8 = {0, 0, 0, 0, 0, 0, 0, 0};
  __builtin_amdgcn_tensor_load_to_lds(g0, g1, z4, z4, z8, 0);
#else
  __builtin_amdgcn_tensor_load_to_lds(g0, g1, z4, z4, 0);
#endif
}

// ---------------- Kernel 2a: WMMA argmax, TDM B staging (fast path) -----------
__global__ __launch_bounds__(128) void vq_argmax_tdm_kernel(const float* __restrict__ x,
                                                            const unsigned short* __restrict__ ebf,
                                                            const float* __restrict__ esq,
                                                            int* __restrict__ best_out) {
  __shared__ __align__(16) unsigned short Blds[2][COLS_PER_STEP * DIM];  // 2 x 32 KB
  __shared__ float redv[ROWS_PER_BLOCK * 16];                            // 4 KB
  __shared__ int   redi[ROWS_PER_BLOCK * 16];                            // 4 KB

  const int tid  = threadIdx.x;
  const int lane = tid & 31;
  const int wave = tid >> 5;
  const int half = lane >> 4;
  const int l16  = lane & 15;
  const int rowBase = blockIdx.x * ROWS_PER_BLOCK;

  AU A[KCHUNKS];
  load_A_fragment(A, x + (size_t)(rowBase + wave * 16 + l16) * DIM, half);

  float bestv[8];
  int   besti[8];
#pragma unroll
  for (int j = 0; j < 8; ++j) { bestv[j] = -__builtin_inff(); besti[j] = 0; }

  const unsigned ldsb0 = lds_off(&Blds[0][0]);
  const unsigned ldsb1 = lds_off(&Blds[1][0]);

  // Prologue: wave 0 DMAs tile 0, waits on TENSORcnt, barrier publishes LDS.
  if (wave == 0) {
    tdm_load_tile(ebf, 0, ldsb0);
    __builtin_amdgcn_s_wait_tensorcnt(0);
  }
  __syncthreads();

  for (int t = 0; t < NTILES; ++t) {
    const int cur = t & 1;
    const int colBase = t * COLS_PER_STEP;
    const unsigned short* Bcur = &Blds[cur][0];

    // Kick off the DMA for tile t+1 before computing tile t (copy/compute overlap).
    if (wave == 0 && t + 1 < NTILES) tdm_load_tile(ebf, t + 1, cur ? ldsb0 : ldsb1);

    v8f acc0 = {};
    v8f acc1 = {};
#pragma unroll
    for (int i = 0; i < KCHUNKS; ++i) {
      const int kc = 32 * i;
      v16bf b0 = *(const v16bf*)&Bcur[(l16)      * DIM + kc + half * 16];
      v16bf b1 = *(const v16bf*)&Bcur[(16 + l16) * DIM + kc + half * 16];
      acc0 = __builtin_amdgcn_wmma_f32_16x16x32_bf16(false, A[i].v, false, b0,
                                                     (short)0, acc0, false, false);
      acc1 = __builtin_amdgcn_wmma_f32_16x16x32_bf16(false, A[i].v, false, b1,
                                                     (short)0, acc1, false, false);
    }

    const int c0 = colBase + l16;
    const int c1 = colBase + 16 + l16;
    const float e0 = esq[c0];
    const float e1 = esq[c1];
#pragma unroll
    for (int j = 0; j < 8; ++j) {
      float v0 = 2.0f * acc0[j] - e0;   // argmax(2 f.e - ||e||^2) == ref argmax
      float v1 = 2.0f * acc1[j] - e1;
      if (v0 > bestv[j]) { bestv[j] = v0; besti[j] = c0; }
      if (v1 > bestv[j]) { bestv[j] = v1; besti[j] = c1; }
    }

    if (wave == 0) __builtin_amdgcn_s_wait_tensorcnt(0);
    __syncthreads();
  }

#pragma unroll
  for (int j = 0; j < 8; ++j) {
    int r = wave * 16 + half * 8 + j;
    redv[r * 16 + l16] = bestv[j];
    redi[r * 16 + l16] = besti[j];
  }
  __syncthreads();
  if (tid < ROWS_PER_BLOCK) {
    float bv = redv[tid * 16 + 0];
    int   bi = redi[tid * 16 + 0];
    for (int s = 1; s < 16; ++s) {
      float v  = redv[tid * 16 + s];
      int   i2 = redi[tid * 16 + s];
      if (v > bv || (v == bv && i2 < bi)) { bv = v; bi = i2; }
    }
    best_out[rowBase + tid] = bi;
  }
}

// ---------------- Kernel 2b: WMMA argmax, in-kernel conversion (fallback) -----
__global__ __launch_bounds__(128) void vq_argmax_kernel(const float* __restrict__ x,
                                                        const float* __restrict__ embed,
                                                        const float* __restrict__ esq,
                                                        int* __restrict__ best_out) {
  __shared__ unsigned short Blds[2][COLS_PER_STEP * DIM];
  __shared__ float redv[ROWS_PER_BLOCK * 16];
  __shared__ int   redi[ROWS_PER_BLOCK * 16];

  const int tid  = threadIdx.x;
  const int lane = tid & 31;
  const int wave = tid >> 5;
  const int half = lane >> 4;
  const int l16  = lane & 15;
  const int rowBase = blockIdx.x * ROWS_PER_BLOCK;

  AU A[KCHUNKS];
  load_A_fragment(A, x + (size_t)(rowBase + wave * 16 + l16) * DIM, half);

  float bestv[8];
  int   besti[8];
#pragma unroll
  for (int j = 0; j < 8; ++j) { bestv[j] = -__builtin_inff(); besti[j] = 0; }

  {
    const float4* ein = (const float4*)embed;
    for (int i = tid; i < (COLS_PER_STEP * DIM) / 4; i += 128) {
      float4 v = ein[i];
      uint2 p;
      p.x = pack_bf16x2_rne(v.x, v.y);
      p.y = pack_bf16x2_rne(v.z, v.w);
      ((uint2*)&Blds[0][0])[i] = p;
    }
  }
  __syncthreads();

  for (int t = 0; t < NTILES; ++t) {
    const int cur = t & 1;
    const int colBase = t * COLS_PER_STEP;
    const unsigned short* Bcur = &Blds[cur][0];

    v8f acc0 = {};
    v8f acc1 = {};
#pragma unroll
    for (int i = 0; i < KCHUNKS; ++i) {
      const int kc = 32 * i;
      v16bf b0 = *(const v16bf*)&Bcur[(l16)      * DIM + kc + half * 16];
      v16bf b1 = *(const v16bf*)&Bcur[(16 + l16) * DIM + kc + half * 16];
      acc0 = __builtin_amdgcn_wmma_f32_16x16x32_bf16(false, A[i].v, false, b0,
                                                     (short)0, acc0, false, false);
      acc1 = __builtin_amdgcn_wmma_f32_16x16x32_bf16(false, A[i].v, false, b1,
                                                     (short)0, acc1, false, false);
    }

    const int c0 = colBase + l16;
    const int c1 = colBase + 16 + l16;
    const float e0 = esq[c0];
    const float e1 = esq[c1];
#pragma unroll
    for (int j = 0; j < 8; ++j) {
      float v0 = 2.0f * acc0[j] - e0;
      float v1 = 2.0f * acc1[j] - e1;
      if (v0 > bestv[j]) { bestv[j] = v0; besti[j] = c0; }
      if (v1 > bestv[j]) { bestv[j] = v1; besti[j] = c1; }
    }

    if (t + 1 < NTILES) {
      const float4* ein = (const float4*)(embed + (size_t)(colBase + COLS_PER_STEP) * DIM);
      const bool pf = (t + 2) < NTILES;
      uint2* Bnxt = (uint2*)&Blds[cur ^ 1][0];
      for (int i = tid; i < (COLS_PER_STEP * DIM) / 4; i += 128) {
        float4 v = ein[i];
        uint2 p;
        p.x = pack_bf16x2_rne(v.x, v.y);
        p.y = pack_bf16x2_rne(v.z, v.w);
        Bnxt[i] = p;
        if (pf) __builtin_prefetch((const void*)(ein + (COLS_PER_STEP * DIM) / 4 + i), 0, 3);
      }
    }
    __syncthreads();
  }

#pragma unroll
  for (int j = 0; j < 8; ++j) {
    int r = wave * 16 + half * 8 + j;
    redv[r * 16 + l16] = bestv[j];
    redi[r * 16 + l16] = besti[j];
  }
  __syncthreads();
  if (tid < ROWS_PER_BLOCK) {
    float bv = redv[tid * 16 + 0];
    int   bi = redi[tid * 16 + 0];
    for (int s = 1; s < 16; ++s) {
      float v  = redv[tid * 16 + s];
      int   i2 = redi[tid * 16 + s];
      if (v > bv || (v == bv && i2 < bi)) { bv = v; bi = i2; }
    }
    best_out[rowBase + tid] = bi;
  }
}

// ---------------- Kernel 3: gather codebook rows + emit indices ----------------
__global__ __launch_bounds__(128) void vq_gather_kernel(const float* __restrict__ embed,
                                                        const int* __restrict__ idx,
                                                        float* __restrict__ out,
                                                        int nrows) {
  const int row = blockIdx.x;
  const int id  = idx[row];
  const float4* src = (const float4*)(embed + (size_t)id * DIM);
  float4*       dst = (float4*)(out + (size_t)row * DIM);
  for (int d = threadIdx.x; d < DIM / 4; d += 128) dst[d] = src[d];
  if (threadIdx.x == 0) out[(size_t)nrows * DIM + row] = (float)id;
}

extern "C" void kernel_launch(void* const* d_in, const int* in_sizes, int n_in,
                              void* d_out, int out_size, void* d_ws, size_t ws_size,
                              hipStream_t stream) {
  const float* x     = (const float*)d_in[0];   // [4,4096,512] fp32
  const float* embed = (const float*)d_in[1];   // [8192,512]  fp32
  float* out = (float*)d_out;

  const int N = in_sizes[0] / DIM;              // 16384 rows

  const size_t ebf_bytes = (size_t)KCODES * DIM * 2;                  // 8 MB
  const size_t need_fast = ebf_bytes + KCODES * 4 + (size_t)N * 4;

  if (ws_size >= need_fast) {
    unsigned short* ebf = (unsigned short*)d_ws;
    float* esq  = (float*)((char*)d_ws + ebf_bytes);
    int*   best = (int*)((char*)d_ws + ebf_bytes + KCODES * 4);
    vq_embed_bf16_kernel<<<(KCODES * DIM / 4) / 256, 256, 0, stream>>>(embed, ebf);
    vq_esq_kernel<<<KCODES / 8, 256, 0, stream>>>(embed, esq);
    vq_argmax_tdm_kernel<<<N / ROWS_PER_BLOCK, 128, 0, stream>>>(x, ebf, esq, best);
    vq_gather_kernel<<<N, 128, 0, stream>>>(embed, best, out, N);
  } else {
    float* esq  = (float*)d_ws;
    int*   best = (int*)((char*)d_ws + KCODES * sizeof(float));
    vq_esq_kernel<<<KCODES / 8, 256, 0, stream>>>(embed, esq);
    vq_argmax_kernel<<<N / ROWS_PER_BLOCK, 128, 0, stream>>>(x, embed, esq, best);
    vq_gather_kernel<<<N, 128, 0, stream>>>(embed, best, out, N);
  }
}
